// AllRecDrugModel_20366734917813
// MI455X (gfx1250) — matrix operand order repaired and checked by reading, not compiled, not run on hardware
//
#include <hip/hip_runtime.h>
#include <hip/hip_bf16.h>
#include <math.h>

// ---------------------------------------------------------------------------
// Model dims (fixed by the reference)
// ---------------------------------------------------------------------------
#define V0 8000
#define V1 4000
#define V2 4096
#define E_DIM 512
#define T_VIS 64
#define L_CODES 48

typedef __attribute__((ext_vector_type(16))) __bf16 v16bf;
typedef __attribute__((ext_vector_type(2)))  __bf16 v2bf;
typedef __attribute__((ext_vector_type(8)))  float  v8f;

__device__ __forceinline__ float sigmoidf_(float x) {
    return 1.0f / (1.0f + __expf(-x));
}

// fp32 pair -> packed bf16 dword (prefer the single-op packed convert)
__device__ __forceinline__ unsigned pack2bf_(float lo, float hi) {
#if __has_builtin(__builtin_amdgcn_cvt_pk_bf16_f32)
    union { v2bf v; unsigned u; } cv;
    cv.v = __builtin_amdgcn_cvt_pk_bf16_f32(lo, hi);
    return cv.u;
#else
    union { unsigned u; __bf16 h[2]; } cv;
    cv.h[0] = (__bf16)lo;
    cv.h[1] = (__bf16)hi;
    return cv.u;
#endif
}

// ---------------------------------------------------------------------------
// 1) Embedding sum-pool:  out[t, :] = sum_l emb[codes[t,l], :]
// ---------------------------------------------------------------------------
__global__ void embed_sum_kernel(const int* __restrict__ codes,
                                 const float* __restrict__ emb,
                                 float* __restrict__ out) {
    const int t = blockIdx.x;
    const int e0 = threadIdx.x;          // 0..255
    float acc0 = 0.f, acc1 = 0.f;
    for (int l = 0; l < L_CODES; ++l) {
        const int c = codes[t * L_CODES + l];
        const float* row = emb + (size_t)c * E_DIM;
        acc0 += row[e0];
        acc1 += row[e0 + 256];
    }
    out[t * E_DIM + e0]       = acc0;
    out[t * E_DIM + e0 + 256] = acc1;
}

// ---------------------------------------------------------------------------
// 2) Aggregation gate + history mask. 1 block, 64 threads (one per visit).
// ---------------------------------------------------------------------------
__global__ void gate_mask_kernel(const float* __restrict__ i1,
                                 const float* __restrict__ i2,
                                 const float* __restrict__ h1W,  // [32,1024]
                                 const float* __restrict__ h1b,  // [32]
                                 const float* __restrict__ gW,   // [1,32]
                                 const float* __restrict__ gb,   // [1]
                                 int* __restrict__ mask) {
    __shared__ float gate_s[T_VIS];
    const int t = threadIdx.x;           // 0..63
    float g = gb[0];
    for (int u = 0; u < 32; ++u) {
        float acc = h1b[u];
        const float* w = h1W + u * (2 * E_DIM);
        const float* a = i1 + t * E_DIM;
        const float* b = i2 + t * E_DIM;
        for (int k = 0; k < E_DIM; ++k) {
            acc += a[k] * w[k] + b[k] * w[E_DIM + k];
        }
        acc = fmaxf(acc, 0.f);
        g += acc * gW[u];
    }
    gate_s[t] = sigmoidf_(g);
    __syncthreads();
    const float last = gate_s[T_VIS - 1];
    mask[t] = (t == T_VIS - 1) || (fabsf(last - gate_s[t]) <= 0.05f);
}

// ---------------------------------------------------------------------------
// 3) Masked GRU over T steps. grid=2 (one block per GRU), 512 threads.
// ---------------------------------------------------------------------------
__global__ void gru_kernel(const float* __restrict__ i1, const float* __restrict__ i2,
                           const int* __restrict__ mask,
                           const float* __restrict__ W1ih, const float* __restrict__ W1hh,
                           const float* __restrict__ b1ih, const float* __restrict__ b1hh,
                           const float* __restrict__ W2ih, const float* __restrict__ W2hh,
                           const float* __restrict__ b2ih, const float* __restrict__ b2hh,
                           float* __restrict__ h1o, float* __restrict__ h2o) {
    const bool second = (blockIdx.x == 1);
    const float* xseq = second ? i2 : i1;
    const float* Wih  = second ? W2ih : W1ih;
    const float* Whh  = second ? W2hh : W1hh;
    const float* bih  = second ? b2ih : b1ih;
    const float* bhh  = second ? b2hh : b1hh;
    float* hout       = second ? h2o  : h1o;

    __shared__ float xs[E_DIM];
    __shared__ float hs[E_DIM];
    const int j = threadIdx.x;
    hs[j] = 0.f;

    const float bir = bih[j], biz = bih[j + E_DIM], bin = bih[j + 2 * E_DIM];
    const float bhr = bhh[j], bhz = bhh[j + E_DIM], bhn = bhh[j + 2 * E_DIM];
    const float* wir = Wih + (size_t)j * E_DIM;
    const float* wiz = Wih + (size_t)(j + E_DIM) * E_DIM;
    const float* win = Wih + (size_t)(j + 2 * E_DIM) * E_DIM;
    const float* whr = Whh + (size_t)j * E_DIM;
    const float* whz = Whh + (size_t)(j + E_DIM) * E_DIM;
    const float* whn = Whh + (size_t)(j + 2 * E_DIM) * E_DIM;

    for (int t = 0; t < T_VIS; ++t) {
        xs[j] = xseq[t * E_DIM + j];
        __syncthreads();
        float gir = bir, giz = biz, gin = bin;
        float ghr = bhr, ghz = bhz, ghn = bhn;
        #pragma unroll 4
        for (int k = 0; k < E_DIM; ++k) {
            const float xv = xs[k];
            const float hv = hs[k];
            gir = fmaf(wir[k], xv, gir);
            giz = fmaf(wiz[k], xv, giz);
            gin = fmaf(win[k], xv, gin);
            ghr = fmaf(whr[k], hv, ghr);
            ghz = fmaf(whz[k], hv, ghz);
            ghn = fmaf(whn[k], hv, ghn);
        }
        const float r = sigmoidf_(gir + ghr);
        const float z = sigmoidf_(giz + ghz);
        const float n = tanhf(gin + r * ghn);
        const float hold = hs[j];
        const float hnew = (1.f - z) * n + z * hold;
        __syncthreads();
        hs[j] = mask[t] ? hnew : hold;
    }
    __syncthreads();
    hout[j] = hs[j];
}

// ---------------------------------------------------------------------------
// 4) query = relu(concat(h1,h2)) @ query_W.T + query_b.  1 block, 512 threads.
// ---------------------------------------------------------------------------
__global__ void query_kernel(const float* __restrict__ h1, const float* __restrict__ h2,
                             const float* __restrict__ qW,  // [512,1024]
                             const float* __restrict__ qb,  // [512]
                             float* __restrict__ query) {
    __shared__ float qin[2 * E_DIM];
    const int j = threadIdx.x;
    qin[j]          = fmaxf(h1[j], 0.f);
    qin[j + E_DIM]  = fmaxf(h2[j], 0.f);
    __syncthreads();
    float acc = qb[j];
    const float* w = qW + (size_t)j * (2 * E_DIM);
    for (int k = 0; k < 2 * E_DIM; ++k) acc = fmaf(qin[k], w[k], acc);
    query[j] = acc;
}

// ---------------------------------------------------------------------------
// 5) Grid-stride copy (drug <- emb2)
// ---------------------------------------------------------------------------
__global__ void copy_kernel(const float* __restrict__ src, float* __restrict__ dst, int n) {
    for (int i = blockIdx.x * blockDim.x + threadIdx.x; i < n; i += gridDim.x * blockDim.x)
        dst[i] = src[i];
}

// ---------------------------------------------------------------------------
// 6) bf16 WMMA GEMM:  C[M,N] (+)= epilogue(A[M,K] @ B[K,N])
//    Fragment-major bf16 LDS staging, double-buffered, 1 barrier / K-tile.
//    MODE 0: C = X ; MODE 1: C = relu(X+bias[n]) ; MODE 2: C += sc*(X+bias[n])
//    Block 256 thr (8 waves); tile 64x64x64; wave: tileM=w&3, tileN pair w>>2.
// ---------------------------------------------------------------------------
#define BLK_MK 64

template <int MODE>
__global__ __launch_bounds__(256)
void gemm_bf16_wmma(const float* __restrict__ A, const float* __restrict__ B,
                    const float* __restrict__ bias, const float* __restrict__ scale_ptr,
                    float scale_mul, float* __restrict__ C, int M, int N, int K) {
    __shared__ v16bf SH[2 * 512];   // 32 KB: 2 buffers x (A:256 | B:256) v16bf

    const int tid  = threadIdx.x;
    const int wave = tid >> 5;
    const int lane = tid & 31;
    const int tileM = wave & 3;
    const int tN0   = (wave >> 2) * 2;
    const int tN1   = tN0 + 1;
    const int mBase = blockIdx.y * BLK_MK;
    const int nBase = blockIdx.x * BLK_MK;

    v8f c0 = {}, c1 = {};

    auto stage = [&](int buf, int kt) {
        unsigned* As32 = (unsigned*)(SH + buf * 512);
        unsigned* Bs32 = (unsigned*)(SH + buf * 512 + 256);
        #pragma unroll
        for (int i = 0; i < 4; ++i) {
            const int p = tid + i * 256;
            const int r = p >> 4;
            const int c = (p & 15) << 2;
            const float4 av = *(const float4*)(A + (size_t)(mBase + r) * K + kt + c);
            const int kstep = c >> 5;
            const int c32   = c & 31;
            const int lf    = (r & 15) | (((c32 >> 3) & 1) << 4);
            const int tM    = r >> 4;
            const int vv    = ((c32 >> 4) << 2) + ((c32 & 7) >> 1);
            uint2 w;
            w.x = pack2bf_(av.x, av.y);
            w.y = pack2bf_(av.z, av.w);
            *(uint2*)&As32[(((kstep << 2) + tM) * 32 + lf) * 8 + vv] = w;
        }
        #pragma unroll
        for (int i = 0; i < 4; ++i) {
            const int p = tid + i * 256;
            const int n = p & 63;
            const int k = (p >> 6) << 2;
            const float* bp = B + (size_t)(kt + k) * N + nBase + n;
            const float b0v = bp[0];
            const float b1v = bp[(size_t)N];
            const float b2v = bp[2 * (size_t)N];
            const float b3v = bp[3 * (size_t)N];
            const int kstep = k >> 5;
            const int k32   = k & 31;
            const int lf    = (n & 15) | (((k32 >> 3) & 1) << 4);
            const int tN    = n >> 4;
            const int vv    = ((k32 >> 4) << 2) + ((k32 & 7) >> 1);
            uint2 w;
            w.x = pack2bf_(b0v, b1v);
            w.y = pack2bf_(b2v, b3v);
            *(uint2*)&Bs32[(((kstep << 2) + tN) * 32 + lf) * 8 + vv] = w;
        }
    };

    stage(0, 0);
    const int iters = K >> 6;
    for (int it = 0; it < iters; ++it) {
        __syncthreads();
        if (it + 1 < iters) stage((it + 1) & 1, (it + 1) << 6);

        const v16bf* Af = SH + (it & 1) * 512;
        const v16bf* Bf = Af + 256;
        #pragma unroll
        for (int ks = 0; ks < 2; ++ks) {
            const v16bf a  = Af[((ks << 2) + tileM) * 32 + lane];
            const v16bf b0 = Bf[((ks << 2) + tN0) * 32 + lane];
            const v16bf b1 = Bf[((ks << 2) + tN1) * 32 + lane];
            c0 = __builtin_amdgcn_wmma_f32_16x16x32_bf16(
                     false, a, false, b0, (short)0, c0, false, false);
            c1 = __builtin_amdgcn_wmma_f32_16x16x32_bf16(
                     false, a, false, b1, (short)0, c1, false, false);
        }
    }

    const int col0 = nBase + tN0 * 16 + (lane & 15);
    const int col1 = col0 + 16;
    const int rowBase = mBase + tileM * 16 + (lane >> 4) * 8;
    float sc = scale_mul;
    if (MODE == 2 && scale_ptr != nullptr) sc *= scale_ptr[0];
    #pragma unroll
    for (int r = 0; r < 8; ++r) {
        const size_t rb = (size_t)(rowBase + r) * N;
        if (MODE == 0) {
            C[rb + col0] = c0[r];
            C[rb + col1] = c1[r];
        } else if (MODE == 1) {
            C[rb + col0] = fmaxf(c0[r] + bias[col0], 0.f);
            C[rb + col1] = fmaxf(c1[r] + bias[col1], 0.f);
        } else {
            C[rb + col0] += sc * (c0[r] + bias[col0]);
            C[rb + col1] += sc * (c1[r] + bias[col1]);
        }
    }
}

// ---------------------------------------------------------------------------
// 7) Attention scores: one wave per row.
// ---------------------------------------------------------------------------
__global__ void scores_kernel(const float* __restrict__ query,
                              const float* __restrict__ drug,
                              float* __restrict__ scores) {
    const int wave = threadIdx.x >> 5;
    const int lane = threadIdx.x & 31;
    const int v = blockIdx.x * 4 + wave;
    const float* row = drug + (size_t)v * E_DIM;
    float acc = 0.f;
    for (int k = lane; k < E_DIM; k += 32) acc = fmaf(query[k], row[k], acc);
    #pragma unroll
    for (int off = 16; off > 0; off >>= 1) acc += __shfl_down(acc, off);
    if (lane == 0) scores[v] = acc;
}

// ---------------------------------------------------------------------------
// 8) In-place softmax over V2 scores. 1 block, 1024 threads.
// ---------------------------------------------------------------------------
__global__ void softmax_kernel(float* __restrict__ s, int n) {
    __shared__ float red[1024];
    const int tid = threadIdx.x;
    float m = -3.4e38f;
    for (int i = tid; i < n; i += 1024) m = fmaxf(m, s[i]);
    red[tid] = m; __syncthreads();
    for (int off = 512; off > 0; off >>= 1) {
        if (tid < off) red[tid] = fmaxf(red[tid], red[tid + off]);
        __syncthreads();
    }
    const float mx = red[0]; __syncthreads();
    float sum = 0.f;
    for (int i = tid; i < n; i += 1024) sum += __expf(s[i] - mx);
    red[tid] = sum; __syncthreads();
    for (int off = 512; off > 0; off >>= 1) {
        if (tid < off) red[tid] += red[tid + off];
        __syncthreads();
    }
    const float inv = 1.f / red[0];
    for (int i = tid; i < n; i += 1024) s[i] = __expf(s[i] - mx) * inv;
}

// ---------------------------------------------------------------------------
// 9) med[j] = sum_v kw[v] * drug[v,j]   -> writes final[512 + j]
// ---------------------------------------------------------------------------
__global__ void med_kernel(const float* __restrict__ kw, const float* __restrict__ drug,
                           float* __restrict__ finalv) {
    const int j = threadIdx.x;
    float acc = 0.f;
    for (int v = 0; v < V2; ++v) acc = fmaf(kw[v], drug[(size_t)v * E_DIM + j], acc);
    finalv[E_DIM + j] = acc;
}

// ---------------------------------------------------------------------------
// 10) LayerNorm of query -> final[0:512]. 1 block, 512 threads.
// ---------------------------------------------------------------------------
__global__ void ln_kernel(const float* __restrict__ query,
                          const float* __restrict__ gamma, const float* __restrict__ beta,
                          float* __restrict__ finalv) {
    __shared__ float red[512];
    const int j = threadIdx.x;
    const float q = query[j];
    red[j] = q; __syncthreads();
    for (int off = 256; off > 0; off >>= 1) {
        if (j < off) red[j] += red[j + off];
        __syncthreads();
    }
    const float mu = red[0] * (1.f / E_DIM); __syncthreads();
    const float d = q - mu;
    red[j] = d * d; __syncthreads();
    for (int off = 256; off > 0; off >>= 1) {
        if (j < off) red[j] += red[j + off];
        __syncthreads();
    }
    const float var = red[0] * (1.f / E_DIM);
    finalv[j] = d * rsqrtf(var + 1e-5f) * gamma[j] + beta[j];
}

// ---------------------------------------------------------------------------
// 11) result[c] = cls_b[c] + <relu(final), cls_W[c,:]>. One wave per row.
// ---------------------------------------------------------------------------
__global__ void cls_kernel(const float* __restrict__ finalv,
                           const float* __restrict__ clsW, const float* __restrict__ clsb,
                           float* __restrict__ result) {
    const int wave = threadIdx.x >> 5;
    const int lane = threadIdx.x & 31;
    const int c = blockIdx.x * 4 + wave;
    const float* row = clsW + (size_t)c * (2 * E_DIM);
    float acc = 0.f;
    for (int k = lane; k < 2 * E_DIM; k += 32)
        acc = fmaf(fmaxf(finalv[k], 0.f), row[k], acc);
    #pragma unroll
    for (int off = 16; off > 0; off >>= 1) acc += __shfl_down(acc, off);
    if (lane == 0) result[c] = acc + clsb[c];
}

__global__ void zero1_kernel(float* p) { *p = 0.f; }

// ---------------------------------------------------------------------------
// 12) batch_neg += 0.0005 * sum_ij sig(r_i) sig(r_j) ddi[i,j]. grid=V2, 256 thr.
// ---------------------------------------------------------------------------
__global__ void ddi_loss_kernel(const float* __restrict__ result,
                                const float* __restrict__ ddi,
                                float* __restrict__ out_scalar) {
    __shared__ float red[256];
    const int i = blockIdx.x;
    const int tid = threadIdx.x;
    const float* row = ddi + (size_t)i * V2;
    float acc = 0.f;
    for (int jj = tid; jj < V2; jj += 256)
        acc = fmaf(sigmoidf_(result[jj]), row[jj], acc);
    red[tid] = acc; __syncthreads();
    for (int off = 128; off > 0; off >>= 1) {
        if (tid < off) red[tid] += red[tid + off];
        __syncthreads();
    }
    if (tid == 0)
        atomicAdd(out_scalar, 0.0005f * sigmoidf_(result[i]) * red[0]);
}

// ---------------------------------------------------------------------------
// kernel_launch
// ---------------------------------------------------------------------------
extern "C" void kernel_launch(void* const* d_in, const int* in_sizes, int n_in,
                              void* d_out, int out_size, void* d_ws, size_t ws_size,
                              hipStream_t stream) {
    const int*   diag_codes = (const int*)  d_in[0];
    const int*   proc_codes = (const int*)  d_in[1];
    const float* emb0       = (const float*)d_in[2];
    const float* emb1       = (const float*)d_in[3];
    const float* emb2       = (const float*)d_in[4];
    const float* g1Wih = (const float*)d_in[5],  *g1Whh = (const float*)d_in[6];
    const float* g1bih = (const float*)d_in[7],  *g1bhh = (const float*)d_in[8];
    const float* g2Wih = (const float*)d_in[9],  *g2Whh = (const float*)d_in[10];
    const float* g2bih = (const float*)d_in[11], *g2bhh = (const float*)d_in[12];
    const float* query_W = (const float*)d_in[13], *query_b = (const float*)d_in[14];
    const float* ph1W = (const float*)d_in[15], *ph1b = (const float*)d_in[16];
    const float* pgW  = (const float*)d_in[17], *pgb  = (const float*)d_in[18];
    const float* clsW = (const float*)d_in[19], *clsb = (const float*)d_in[20];
    const float* ln_g = (const float*)d_in[21], *ln_b = (const float*)d_in[22];
    const float* eW1 = (const float*)d_in[23], *eb1 = (const float*)d_in[24];
    const float* eW2 = (const float*)d_in[25], *eb2 = (const float*)d_in[26];
    const float* dW1 = (const float*)d_in[27], *db1 = (const float*)d_in[28];
    const float* dW2 = (const float*)d_in[29], *db2 = (const float*)d_in[30];
    const float* inter1   = (const float*)d_in[31];
    const float* ehr_adjn = (const float*)d_in[32];
    const float* ddi_adjn = (const float*)d_in[33];
    const float* ddi_adj  = (const float*)d_in[34];

    float* ws = (float*)d_ws;
    float* i1     = ws;
    float* i2     = i1 + T_VIS * E_DIM;
    int*   mask   = (int*)(i2 + T_VIS * E_DIM);
    float* h1     = (float*)(mask + T_VIS);
    float* h2     = h1 + E_DIM;
    float* query  = h2 + E_DIM;
    float* finalv = query + E_DIM;
    float* kw     = finalv + 2 * E_DIM;
    float* Hbuf   = kw + V2;
    float* Mbuf   = Hbuf + (size_t)V2 * E_DIM;
    float* drug   = Mbuf + (size_t)V2 * E_DIM;

    float* out = (float*)d_out;

    embed_sum_kernel<<<T_VIS, 256, 0, stream>>>(diag_codes, emb0, i1);
    embed_sum_kernel<<<T_VIS, 256, 0, stream>>>(proc_codes, emb1, i2);
    gate_mask_kernel<<<1, T_VIS, 0, stream>>>(i1, i2, ph1W, ph1b, pgW, pgb, mask);
    gru_kernel<<<2, E_DIM, 0, stream>>>(i1, i2, mask,
                                        g1Wih, g1Whh, g1bih, g1bhh,
                                        g2Wih, g2Whh, g2bih, g2bhh, h1, h2);
    query_kernel<<<1, E_DIM, 0, stream>>>(h1, h2, query_W, query_b, query);

    copy_kernel<<<2048, 256, 0, stream>>>(emb2, drug, V2 * E_DIM);

    const dim3 ggrid(E_DIM / BLK_MK, V2 / BLK_MK);
    gemm_bf16_wmma<1><<<ggrid, 256, 0, stream>>>(ehr_adjn, eW1, eb1, nullptr,  1.f, Hbuf, V2, E_DIM, V2);
    gemm_bf16_wmma<0><<<ggrid, 256, 0, stream>>>(Hbuf, eW2, nullptr, nullptr,  1.f, Mbuf, V2, E_DIM, E_DIM);
    gemm_bf16_wmma<2><<<ggrid, 256, 0, stream>>>(ehr_adjn, Mbuf, eb2, nullptr, 1.f, drug, V2, E_DIM, V2);
    gemm_bf16_wmma<1><<<ggrid, 256, 0, stream>>>(ddi_adjn, dW1, db1, nullptr,  1.f, Hbuf, V2, E_DIM, V2);
    gemm_bf16_wmma<0><<<ggrid, 256, 0, stream>>>(Hbuf, dW2, nullptr, nullptr,  1.f, Mbuf, V2, E_DIM, E_DIM);
    gemm_bf16_wmma<2><<<ggrid, 256, 0, stream>>>(ddi_adjn, Mbuf, db2, inter1, -1.f, drug, V2, E_DIM, V2);

    scores_kernel<<<V2 / 4, 128, 0, stream>>>(query, drug, kw);
    softmax_kernel<<<1, 1024, 0, stream>>>(kw, V2);
    med_kernel<<<1, E_DIM, 0, stream>>>(kw, drug, finalv);
    ln_kernel<<<1, E_DIM, 0, stream>>>(query, ln_g, ln_b, finalv);

    cls_kernel<<<V2 / 4, 128, 0, stream>>>(finalv, clsW, clsb, out);
    zero1_kernel<<<1, 1, 0, stream>>>(out + V2);
    ddi_loss_kernel<<<V2, 256, 0, stream>>>(out, ddi_adj, out + V2);

    (void)in_sizes; (void)n_in; (void)out_size; (void)ws_size;
}
